// GRACE_23630910063292
// MI455X (gfx1250) — matrix operand order repaired and checked by reading, not compiled, not run on hardware
//
#include <hip/hip_runtime.h>

// ---------------------------------------------------------------------------
// GCN backbone for MI455X (gfx1250, wave32, WMMA)
//   out = D^-1/2 (A+I) D^-1/2 (X W) + b   , two layers, two graphs.
// Dense transform: v_wmma_f32_16x16x32_bf16 with W pre-swizzled in LDS into
// per-lane B-fragment layout (vector ds_load_b128 in the inner loop).
// Edge aggregation: fp32 global atomics (working set fits the 192MB L2).
// ---------------------------------------------------------------------------

typedef __attribute__((ext_vector_type(16))) __bf16 v16bf;
typedef __attribute__((ext_vector_type(8)))  float  v8f;

#define N_NODES 50000
#define N_EDGES 800000
#define DIM     128

// ------------------------------- degree ------------------------------------
__global__ void deg_init_kernel(float* __restrict__ deg, int n) {
    int i = blockIdx.x * blockDim.x + threadIdx.x;
    if (i < n) deg[i] = 1.0f;                       // self-loop
}

__global__ void deg_accum_kernel(const int* __restrict__ dst,
                                 float* __restrict__ deg, int n_edges) {
    int i = blockIdx.x * blockDim.x + threadIdx.x;
    if (i < n_edges) atomicAdd(&deg[dst[i]], 1.0f);
}

__global__ void dinv_kernel(float* __restrict__ deg, int n) {
    int i = blockIdx.x * blockDim.x + threadIdx.x;
    if (i < n) deg[i] = rsqrtf(deg[i]);             // in place: deg -> dinv
}

// ------------------------------- GEMM --------------------------------------
// H[n_rows,128] = (RELU? relu(X) : X)[n_rows,128] @ W[128,128]
// Block: 256 threads = 8 waves; each wave computes 16 rows x 128 cols.
// n_rows must be a multiple of 16 (50000 = 3125*16).
//
// B fragments (32x16 bf16 per ISA 7.12.2 wave32 layout) are pre-packed into
// LDS so each lane's operand is 32 contiguous bytes:
//   entry e = (ks*8 + nt)*32 + lane   holds   b[2v]   = W[ks*32 + half*8 + 2v + (v>=4?8:0)    , nt*16 + mlane]
//                                             b[2v+1] = W[ks*32 + half*8 + 2v + (v>=4?8:0) + 1, nt*16 + mlane]
template <bool RELU>
__global__ __launch_bounds__(256) void gemm_node_kernel(
    const float* __restrict__ X,
    const float* __restrict__ W,     // [K=128, N=128] row-major
    float* __restrict__ H,
    int n_rows)
{
    __shared__ __align__(32) __bf16 sWp[4 * 8 * 32 * 16];   // 32 KB packed fragments

    // ---- one-time pack: 1024 fragment-lane entries, 4 per thread ----
    #pragma unroll
    for (int rep = 0; rep < 4; ++rep) {
        int e = rep * 256 + threadIdx.x;        // 0..1023
        int ks    = e >> 8;                     // K tile (0..3)
        int nt    = (e >> 5) & 7;               // N tile (0..7)
        int lanee = e & 31;
        int mlane = lanee & 15;
        int half  = lanee >> 4;
        int n     = nt * 16 + mlane;
        int kb    = ks * 32 + half * 8;
        v16bf frag;
        #pragma unroll
        for (int v = 0; v < 8; ++v) {
            frag[2 * v]     = (__bf16)W[(kb + 2 * v + (v >= 4 ? 8 : 0))     * DIM + n];
            frag[2 * v + 1] = (__bf16)W[(kb + 2 * v + (v >= 4 ? 8 : 0) + 1) * DIM + n];
        }
        *((v16bf*)sWp + e) = frag;
    }
    __syncthreads();

    const int wave = threadIdx.x >> 5;
    const int lane = threadIdx.x & 31;
    const int row0 = (blockIdx.x * 8 + wave) * 16;
    if (row0 >= n_rows) return;                 // wave-uniform: EXEC stays full

    const int mlane = lane & 15;
    const int half  = lane >> 4;                // 0 or 1

    const float* xrow = X + (size_t)(row0 + mlane) * DIM;

    v8f acc[8] = {};

    #pragma unroll
    for (int ks = 0; ks < 4; ++ks) {            // K step of 32
        const int kbase0 = ks * 32;

        // A fragment: 16x32 bf16. Lane reads two contiguous 8-float runs:
        //   [kbase0 + 8*half, +8) -> VGPR 0..3 ; [kbase0 + 16 + 8*half, +8) -> VGPR 4..7
        float4 s0 = *(const float4*)(xrow + kbase0 + half * 8);
        float4 s1 = *(const float4*)(xrow + kbase0 + half * 8 + 4);
        float4 s2 = *(const float4*)(xrow + kbase0 + 16 + half * 8);
        float4 s3 = *(const float4*)(xrow + kbase0 + 16 + half * 8 + 4);
        if (RELU) {
            s0.x = fmaxf(s0.x, 0.f); s0.y = fmaxf(s0.y, 0.f); s0.z = fmaxf(s0.z, 0.f); s0.w = fmaxf(s0.w, 0.f);
            s1.x = fmaxf(s1.x, 0.f); s1.y = fmaxf(s1.y, 0.f); s1.z = fmaxf(s1.z, 0.f); s1.w = fmaxf(s1.w, 0.f);
            s2.x = fmaxf(s2.x, 0.f); s2.y = fmaxf(s2.y, 0.f); s2.z = fmaxf(s2.z, 0.f); s2.w = fmaxf(s2.w, 0.f);
            s3.x = fmaxf(s3.x, 0.f); s3.y = fmaxf(s3.y, 0.f); s3.z = fmaxf(s3.z, 0.f); s3.w = fmaxf(s3.w, 0.f);
        }
        v16bf a;
        a[0]  = (__bf16)s0.x; a[1]  = (__bf16)s0.y; a[2]  = (__bf16)s0.z; a[3]  = (__bf16)s0.w;
        a[4]  = (__bf16)s1.x; a[5]  = (__bf16)s1.y; a[6]  = (__bf16)s1.z; a[7]  = (__bf16)s1.w;
        a[8]  = (__bf16)s2.x; a[9]  = (__bf16)s2.y; a[10] = (__bf16)s2.z; a[11] = (__bf16)s2.w;
        a[12] = (__bf16)s3.x; a[13] = (__bf16)s3.y; a[14] = (__bf16)s3.z; a[15] = (__bf16)s3.w;

        #pragma unroll
        for (int nt = 0; nt < 8; ++nt) {        // 8 N-tiles of 16
            v16bf b = *((const v16bf*)sWp + ((ks * 8 + nt) * 32 + lane));
            acc[nt] = __builtin_amdgcn_wmma_f32_16x16x32_bf16(
                false, a, false, b, (short)0, acc[nt], false, false);
        }
    }

    // D layout: lane holds N = mlane, rows M = 8*half + v
    #pragma unroll
    for (int nt = 0; nt < 8; ++nt) {
        #pragma unroll
        for (int v = 0; v < 8; ++v) {
            int r = row0 + half * 8 + v;
            H[(size_t)r * DIM + nt * 16 + mlane] = acc[nt][v];
        }
    }
}

// --------------------- self-loop + bias init -------------------------------
// out[i,c] = b[c] + H[i,c] * dinv[i]^2
__global__ void self_init_kernel(const float* __restrict__ H,
                                 const float* __restrict__ dinv,
                                 const float* __restrict__ bias,
                                 float* __restrict__ out, int total)
{
    int i = blockIdx.x * blockDim.x + threadIdx.x;
    if (i >= total) return;
    int r = i >> 7;
    int c = i & (DIM - 1);
    float di = dinv[r];
    out[i] = bias[c] + H[i] * di * di;
}

// --------------------------- edge scatter ----------------------------------
// One wave per edge; each lane owns a float4 of the 128-wide feature row.
__global__ __launch_bounds__(256) void edge_agg_kernel(
    const float* __restrict__ H,
    const int*   __restrict__ src,
    const int*   __restrict__ dst,
    const float* __restrict__ dinv,
    float*       __restrict__ out,
    int n_edges)
{
    int e = blockIdx.x * (blockDim.x >> 5) + (threadIdx.x >> 5);
    if (e >= n_edges) return;
    int lane = threadIdx.x & 31;

    int s = src[e];
    int d = dst[e];
    float norm = dinv[s] * dinv[d];

    const float4* hv = (const float4*)(H + (size_t)s * DIM);
    float4 m = hv[lane];

    float* o = out + (size_t)d * DIM + lane * 4;
    atomicAdd(o + 0, m.x * norm);
    atomicAdd(o + 1, m.y * norm);
    atomicAdd(o + 2, m.z * norm);
    atomicAdd(o + 3, m.w * norm);
}

// ----------------------------- host side -----------------------------------
extern "C" void kernel_launch(void* const* d_in, const int* in_sizes, int n_in,
                              void* d_out, int out_size, void* d_ws, size_t ws_size,
                              hipStream_t stream)
{
    const float* x[2]  = { (const float*)d_in[0], (const float*)d_in[2] };
    const int*   ei[2] = { (const int*)d_in[1],   (const int*)d_in[3]   };
    const float* W0 = (const float*)d_in[4];
    const float* b0 = (const float*)d_in[5];
    const float* W1 = (const float*)d_in[6];
    const float* b1 = (const float*)d_in[7];
    float* out = (float*)d_out;

    // workspace layout (floats): dinv | H | A
    float* ws   = (float*)d_ws;
    float* dinv = ws;                       // N_NODES (reused as deg first)
    float* Hbuf = ws + 50176;               // N_NODES*DIM, 16B-aligned offset
    float* Abuf = Hbuf + (size_t)N_NODES * DIM;

    const int total = N_NODES * DIM;
    const int gemm_blocks = ((N_NODES + 15) / 16 + 7) / 8;     // 391
    const int edge_blocks = (N_EDGES + 7) / 8;                 // wave per edge
    const int nthr = 256;

    for (int g = 0; g < 2; ++g) {
        const int* srcp = ei[g];
        const int* dstp = ei[g] + N_EDGES;
        float* outg = out + (size_t)g * total;

        // degree -> dinv (shared by both layers)
        deg_init_kernel<<<(N_NODES + nthr - 1) / nthr, nthr, 0, stream>>>(dinv, N_NODES);
        deg_accum_kernel<<<(N_EDGES + nthr - 1) / nthr, nthr, 0, stream>>>(dstp, dinv, N_EDGES);
        dinv_kernel<<<(N_NODES + nthr - 1) / nthr, nthr, 0, stream>>>(dinv, N_NODES);

        // ---- layer 0: A = Ahat (X W0) + b0 ----
        gemm_node_kernel<false><<<gemm_blocks, nthr, 0, stream>>>(x[g], W0, Hbuf, N_NODES);
        self_init_kernel<<<(total + nthr - 1) / nthr, nthr, 0, stream>>>(Hbuf, dinv, b0, Abuf, total);
        edge_agg_kernel<<<edge_blocks, nthr, 0, stream>>>(Hbuf, srcp, dstp, dinv, Abuf, N_EDGES);

        // ---- layer 1: out = Ahat (relu(A) W1) + b1 ----
        gemm_node_kernel<true><<<gemm_blocks, nthr, 0, stream>>>(Abuf, W1, Hbuf, N_NODES);
        self_init_kernel<<<(total + nthr - 1) / nthr, nthr, 0, stream>>>(Hbuf, dinv, b1, outg, total);
        edge_agg_kernel<<<edge_blocks, nthr, 0, stream>>>(Hbuf, srcp, dstp, dinv, outg, N_EDGES);
    }
}
